// MAGNN_nc_AC_20856361190124
// MI455X (gfx1250) — matrix operand order repaired and checked by reading, not compiled, not run on hardware
//
#include <hip/hip_runtime.h>
#include <hip/hip_bf16.h>

// MAGNN_nc_AC for MI455X (gfx1250, wave32, WMMA).
// Live computation only: feat_src = feat0@W_fc0+b_fc0 (the returned tf rows),
// plus loss_ac via 8-head attention over 600 drop rows x 1400 keep keys.
// All bounds handling via index clamping -> branch-free WMMA inner loops.

typedef __attribute__((ext_vector_type(16))) _Float16 v16h;
typedef __attribute__((ext_vector_type(2)))  _Float16 v2h;
typedef __attribute__((ext_vector_type(8)))  float    v8f;

#define N0_    2000
#define NKEEP  1400
#define NDROP  600
#define HID    64
#define HEADS  8
#define NADJ   9000
#define KEEPP  1408          // 1400 padded to 32
#define NEGF   (-30000.0f)   // f16-safe "-inf": exp(NEGF - max) flushes to 0

__device__ __forceinline__ v8f wmma_f16(v16h a, v16h b, v8f c) {
  // D = A(16x32 f16) * B(32x16 f16) + C(16x16 f32)
  return __builtin_amdgcn_wmma_f32_16x16x32_f16(false, a, false, b,
                                                (short)0, c, false, false);
}

// Lane layouts (ISA 7.12.2, wave32):
//  A 16x32 f16 : lane l holds row M=l&15; e<8 -> K=(l>>4)*8+e ; e>=8 -> K=16+(l>>4)*8+(e-8)
//                (two contiguous 8-float spans per lane in row-major memory)
//  B 32x16 f16 : lane l holds col N=l&15; element e -> K = (l>>4)*16 + e in this code (A loads use the same order)
//  C 16x16 f32 : reg r, lane l -> M = r + (l>>4)*8, N = l&15

// ---------------------------------------------------------------------------
// GEMM (K compile-time):  C[h] = gather(A[h], aidx) @ B[h] (+bias), N = 64.
// One wave computes one 16x16 tile. All loads unconditional (rows clamped).
// ---------------------------------------------------------------------------
template <int K, bool GATHER, bool BIAS>
__global__ __launch_bounds__(32)
void gemm_wmma(const float* __restrict__ A, const int* __restrict__ aidx,
               long aHeadStride,
               const float* __restrict__ B, long bHeadStride,
               const float* __restrict__ bias,
               float* __restrict__ C, long cHeadStride, int M)
{
  const int lane = threadIdx.x & 31;
  const int hi = lane >> 4, lo = lane & 15;
  const int m0 = blockIdx.x * 16;
  const int n0 = blockIdx.y * 16;
  const int h  = blockIdx.z;
  A += (long)h * aHeadStride;
  B += (long)h * bHeadStride;
  C += (long)h * cHeadStride;

  const int  m   = min(m0 + lo, M - 1);               // clamp: dead rows unstored
  const long arw = (long)(GATHER ? aidx[m] : m) * K;
  const float* ap = A + arw;

  v8f acc = {};
#pragma unroll 2
  for (int k0 = 0; k0 < K; k0 += 32) {
    v16h av, bv;
    const float* a0p = ap + k0 + hi * 16;             // K = k0+hi*16  .. +7
    const float* a1p = a0p + 8;                       // K = k0+hi*16+8.. +7
#pragma unroll
    for (int e = 0; e < 8; ++e) {
      av[e]     = (_Float16)a0p[e];
      av[e + 8] = (_Float16)a1p[e];
    }
    const float* bp = B + (long)(k0 + hi * 16) * HID + n0 + lo;
#pragma unroll
    for (int e = 0; e < 16; ++e)
      bv[e] = (_Float16)bp[(long)e * HID];
    acc = wmma_f16(av, bv, acc);
  }
#pragma unroll
  for (int r = 0; r < 8; ++r) {
    const int mm = m0 + r + hi * 8;
    if (mm < M) {
      float v = acc[r];
      if (BIAS) v += bias[n0 + lo];
      C[(long)mm * HID + n0 + lo] = v;
    }
  }
}

// ---------------------------------------------------------------------------
// Fused attention for 16 drop rows of one head:
//   logits = elu(q @ k^T) masked by adj[drop, keep] (else -inf)
//   out    = softmax(logits) @ feat_src[keep]
// Logits staged in LDS as f16 (16 x 1408 = 45 KB of the 320 KB/WGP).
// ---------------------------------------------------------------------------
__global__ __launch_bounds__(32)
void attn_fused_wmma(const float* __restrict__ qmat,   // HEADS x NDROP x HID
                     const float* __restrict__ kmat,   // HEADS x NKEEP x HID
                     const float* __restrict__ vsrc,   // N0 x HID (= feat_src)
                     const int*  __restrict__ keep_idx,
                     const int*  __restrict__ drop_idx,
                     const float* __restrict__ adj,    // NADJ x NADJ
                     float* __restrict__ headout)      // HEADS x NDROP x HID
{
  __shared__ _Float16 sL[16][KEEPP];
  const int lane = threadIdx.x & 31;
  const int hi = lane >> 4, lo = lane & 15;
  const int i0 = blockIdx.x * 16;
  const int h  = blockIdx.z;
  const float* q  = qmat + (long)h * NDROP * HID;
  const float* km = kmat + (long)h * NKEEP * HID;

  // q tile (16x64) as two f16 A-fragments (contiguous spans, unconditional).
  v16h a0, a1;
  {
    const float* qp = q + (long)min(i0 + lo, NDROP - 1) * HID;
#pragma unroll
    for (int e = 0; e < 8; ++e) {
      a0[e]     = (_Float16)qp[hi * 16 + e];
      a0[e + 8] = (_Float16)qp[hi * 16 + 8 + e];
      a1[e]     = (_Float16)qp[32 + hi * 16 + e];
      a1[e + 8] = (_Float16)qp[32 + hi * 16 + 8 + e];
    }
  }

  // Per-lane adj row bases for the 8 C-rows this lane produces (clamped).
  long adjrow[8];
#pragma unroll
  for (int r = 0; r < 8; ++r)
    adjrow[r] = (long)drop_idx[min(i0 + r + hi * 8, NDROP - 1)] * NADJ;

  // ---- phase 1: masked elu(logits) -> LDS -------------------------------
  for (int c0 = 0; c0 < KEEPP; c0 += 32) {
#pragma unroll
    for (int t = 0; t < 2; ++t) {
      const int  key   = c0 + t * 16 + lo;            // logits column
      const bool keyok = key < NKEEP;
      const int  keyc  = min(key, NKEEP - 1);
      const float* kr  = km + (long)keyc * HID;       // k^T: one kmat row/lane
      v16h b0, b1;
#pragma unroll
      for (int e = 0; e < 16; ++e) {
        b0[e] = (_Float16)kr[hi * 16 + e];
        b1[e] = (_Float16)kr[32 + hi * 16 + e];
      }
      v8f ct = {};
      ct = wmma_f16(a0, b0, ct);
      ct = wmma_f16(a1, b1, ct);
      const int kcol = keep_idx[keyc];
#pragma unroll
      for (int r = 0; r < 8; ++r) {
        const float x  = ct[r];
        const float el = (x > 0.0f) ? x : (__expf(x) - 1.0f);
        const float bv = adj[adjrow[r] + kcol];
        const float logit = (keyok && bv > 0.0f) ? el : NEGF;
        sL[r + hi * 8][key] = (_Float16)logit;
      }
    }
  }
  __syncthreads();

  // ---- phase 2: row max / inverse exp-sum.  Row m = lo; half-wave `hi`
  // scans a contiguous 704-column span as f16x2 (pads are exp-neutral).
  const v2h* rowp = (const v2h*)(&sL[lo][0]);
  float rmax = -3.4e38f;
  for (int n = hi * 352; n < hi * 352 + 352; ++n) {
    const v2h v = rowp[n];
    rmax = fmaxf(rmax, fmaxf((float)v.x, (float)v.y));
  }
  rmax = fmaxf(rmax, __shfl_xor(rmax, 16, 32));
  float rsum = 0.0f;
  for (int n = hi * 352; n < hi * 352 + 352; ++n) {
    const v2h v = rowp[n];
    rsum += __expf((float)v.x - rmax) + __expf((float)v.y - rmax);
  }
  rsum += __shfl_xor(rsum, 16, 32);
  const float rinv = 1.0f / fmaxf(rsum, 1e-30f);
  __syncthreads();

  // ---- phase 3: out = P @ V (V = feat_src rows gathered by keep_idx) -----
  v8f acc[4] = {{}, {}, {}, {}};
  for (int c0 = 0; c0 < KEEPP; c0 += 32) {
    v16h p;
    int vrow[16];
#pragma unroll
    for (int e = 0; e < 8; ++e) {       // A-layout P: one contiguous 16-key LDS span
      p[e]     = (_Float16)(__expf((float)sL[lo][c0 + hi * 16 + e]     - rmax) * rinv);
      p[e + 8] = (_Float16)(__expf((float)sL[lo][c0 + hi * 16 + 8 + e] - rmax) * rinv);
    }
#pragma unroll
    for (int e = 0; e < 16; ++e)        // pad keys: p==0, so clamped V is inert
      vrow[e] = keep_idx[min(c0 + hi * 16 + e, NKEEP - 1)];
#pragma unroll
    for (int t = 0; t < 4; ++t) {
      v16h bv;
#pragma unroll
      for (int e = 0; e < 16; ++e)
        bv[e] = (_Float16)vsrc[(long)vrow[e] * HID + t * 16 + lo];
      acc[t] = wmma_f16(p, bv, acc[t]);
    }
  }
  float* outh = headout + (long)h * NDROP * HID;
#pragma unroll
  for (int t = 0; t < 4; ++t) {
#pragma unroll
    for (int r = 0; r < 8; ++r) {
      const int gm = i0 + r + hi * 8;
      if (gm < NDROP) outh[(long)gm * HID + t * 16 + lo] = acc[t][r];
    }
  }
}

// ---------------------------------------------------------------------------
__global__ void gather_out_kernel(const float* __restrict__ feat_src,
                                  const int* __restrict__ tgt,
                                  float* __restrict__ out, int nrows)
{
  const int i = blockIdx.x * blockDim.x + threadIdx.x;
  if (i < nrows * HID) {
    const int r = i / HID, j = i % HID;
    out[i] = feat_src[(long)tgt[r] * HID + j];
  }
}

__global__ __launch_bounds__(256)
void loss_kernel(const float* __restrict__ feat_src,
                 const float* __restrict__ headout,   // HEADS x NDROP x HID
                 const int* __restrict__ drop_idx,
                 float* __restrict__ out_loss)
{
  __shared__ float red[256];
  const int tid = threadIdx.x;
  float local = 0.0f;
  for (int i = tid; i < NDROP; i += 256) {
    const float* a = feat_src + (long)drop_idx[i] * HID;
    float dot = 0.0f, na2 = 0.0f, nb2 = 0.0f;
    for (int j = 0; j < HID; ++j) {
      float bv = 0.0f;
#pragma unroll
      for (int hh = 0; hh < HEADS; ++hh)
        bv += headout[((long)hh * NDROP + i) * HID + j];
      bv *= (1.0f / HEADS);                       // mean over heads
      const float av = a[j];
      dot += av * bv; na2 += av * av; nb2 += bv * bv;
    }
    const float na = fmaxf(sqrtf(na2), 1e-8f);
    const float nb = fmaxf(sqrtf(nb2), 1e-8f);
    local += dot / (na * nb);
  }
  red[tid] = local;
  __syncthreads();
  for (int s = 128; s > 0; s >>= 1) {
    if (tid < s) red[tid] += red[tid + s];
    __syncthreads();
  }
  if (tid == 0) out_loss[0] = 1.0f - red[0] / (float)NDROP;
}

// ---------------------------------------------------------------------------
extern "C" void kernel_launch(void* const* d_in, const int* in_sizes, int n_in,
                              void* d_out, int out_size, void* d_ws, size_t ws_size,
                              hipStream_t stream)
{
  const float* adj    = (const float*)d_in[0];
  const float* feat0  = (const float*)d_in[1];
  const float* emb    = (const float*)d_in[4];
  const float* W_fc0  = (const float*)d_in[5];
  const float* b_fc0  = (const float*)d_in[6];
  const float* W_att  = (const float*)d_in[11];   // HEADS x 64 x 64
  const float* W2_att = (const float*)d_in[12];   // HEADS x 64 x 64
  const int* keep_idx = (const int*)d_in[16];
  const int* drop_idx = (const int*)d_in[17];
  const int* tgt_idx  = (const int*)d_in[18];
  float* out = (float*)d_out;

  // Workspace layout (floats): ~6 MB total.
  float* ws       = (float*)d_ws;
  float* feat_src = ws;                                    // 2000*64
  float* kmat     = feat_src + (long)N0_ * HID;            // 8*1400*64
  float* qmat     = kmat + (long)HEADS * NKEEP * HID;      // 8*600*64
  float* mhead    = qmat + (long)HEADS * NDROP * HID;      // 8*64*64
  float* headout  = mhead + (long)HEADS * HID * HID;       // 8*600*64

  const dim3 wblk(32);

  // 1) feat_src = feat0 @ W_fc0 + b_fc0          (2000x512 * 512x64)
  gemm_wmma<512, false, true><<<dim3((N0_ + 15) / 16, 4, 1), wblk, 0, stream>>>(
      feat0, nullptr, 0, W_fc0, 0, b_fc0, feat_src, 0, N0_);

  // 2) mhead[h] = W_att[h] @ W2_att[h]           (fold h2@W2 into one projection)
  gemm_wmma<64, false, false><<<dim3(4, 4, HEADS), wblk, 0, stream>>>(
      W_att, nullptr, 64 * 64, W2_att, 64 * 64, nullptr, mhead, 64 * 64, 64);

  // 3) kmat[h] = emb[keep_idx] @ W_att[h]
  gemm_wmma<64, true, false><<<dim3((NKEEP + 15) / 16, 4, HEADS), wblk, 0, stream>>>(
      emb, keep_idx, 0, W_att, 64 * 64, nullptr, kmat, (long)NKEEP * HID, NKEEP);

  // 4) qmat[h] = emb[drop_idx] @ mhead[h]
  gemm_wmma<64, true, false><<<dim3((NDROP + 15) / 16, 4, HEADS), wblk, 0, stream>>>(
      emb, drop_idx, 0, mhead, 64 * 64, nullptr, qmat, (long)NDROP * HID, NDROP);

  // 5) Fused elu/mask/softmax/PV per head (deterministic per-head outputs)
  attn_fused_wmma<<<dim3((NDROP + 15) / 16, 1, HEADS), wblk, 0, stream>>>(
      qmat, kmat, feat_src, keep_idx, drop_idx, adj, headout);

  // 6) tf[target] (= feat_src gathered) and loss
  gather_out_kernel<<<dim3((N0_ * HID + 255) / 256), dim3(256), 0, stream>>>(
      feat_src, tgt_idx, out, N0_);
  loss_kernel<<<dim3(1), dim3(256), 0, stream>>>(
      feat_src, headout, drop_idx, out + (long)N0_ * HID);
}
